// BatchTopKSAE_62199716380829
// MI455X (gfx1250) — matrix operand order repaired and checked by reading, not compiled, not run on hardware
//
#include <hip/hip_runtime.h>

// ---------------------------------------------------------------------------
// BatchTopK SAE forward for MI455X (gfx1250, wave32, WMMA + async-LDS)
//   encoded = x @ enc_w.T + enc_b          [8192, 16384]
//   sparse  = batch_top_k(encoded, 1%)     (top-81 per column over batch dim)
//   decoded = sparse @ dec_w.T + dec_b     [8192, 768]
// Output layout: [decoded (8192*768) | sparse (8192*16384)] f32
// ---------------------------------------------------------------------------

typedef __attribute__((ext_vector_type(16))) __bf16 v16bf;
typedef __attribute__((ext_vector_type(8)))  __bf16 v8bf;
typedef __attribute__((ext_vector_type(8)))  float  v8f;
typedef __attribute__((ext_vector_type(4)))  float  v4f;

#define BATCH      8192
#define IN_DIM     768
#define LATENT     16384
#define TOPK       81      // int(8192 * 0.01)

#define BM 128
#define BN 128
#define BK 32
#define LDSF 36            // f32 LDS row stride (144 B, 16B-aligned)
#define LDB  40            // bf16 LDS row stride (80 B, 16B-aligned)

// Per-lane async copy of 16 bytes global -> LDS (ASYNCcnt-tracked, no VGPRs).
__device__ __forceinline__ void async_b128_to_lds(const float* gptr, float* lptr)
{
    // Low 32 bits of a generic LDS pointer are the LDS byte offset.
    unsigned lds_off = (unsigned)(size_t)lptr;
    asm volatile("global_load_async_to_lds_b128 %0, %1, off"
                 :: "v"(lds_off), "v"(gptr)
                 : "memory");
}

__device__ __forceinline__ void wait_async_all()
{
    asm volatile("s_wait_asynccnt 0x0" ::: "memory");
}

// C[M,N] = A[M,K] * B[N,K]^T + bias[N].  f32 in memory, bf16 WMMA math.
// Pipeline per K-step: async f32 tile (double-buffered) -> one cooperative
// f32->bf16 convert (each element converted exactly once) -> bf16 fragments
// -> 8 WMMAs/wave.  M%128==0, N%128==0, K%32==0.
__global__ __launch_bounds__(256)
void sae_gemm_bf16_wmma(const float* __restrict__ A,
                        const float* __restrict__ B,
                        const float* __restrict__ bias,
                        float* __restrict__ C,
                        int M, int N, int K)
{
    __shared__ __align__(16) float  Asf[2][BM * LDSF];  // async f32 landing pads
    __shared__ __align__(16) float  Bsf[2][BN * LDSF];
    __shared__ __align__(16) __bf16 Asb[BM * LDB];      // converted bf16 tile
    __shared__ __align__(16) __bf16 Bsb[BN * LDB];

    const int t    = threadIdx.x;
    const int lane = t & 31;
    const int wave = t >> 5;
    const int wm   = wave & 3;     // 4 waves along M -> 32 rows each
    const int wn   = wave >> 2;    // 2 waves along N -> 64 cols each

    const int blockM = blockIdx.y * BM;
    const int blockN = blockIdx.x * BN;

    // Async staging coordinates: 2048 b128 transfers per tile pair, 8/thread.
    const int srow0 = t >> 3;           // rows t/8, +32, +64, +96
    const int sc4   = (t & 7) * 4;      // f32 column within the 32-wide tile

    v8f acc[2][4];
#pragma unroll
    for (int i = 0; i < 2; ++i)
#pragma unroll
        for (int j = 0; j < 4; ++j)
            acc[i][j] = (v8f){0.f, 0.f, 0.f, 0.f, 0.f, 0.f, 0.f, 0.f};

    const int mrow = lane & 15;
    const int kha  = (lane >> 4) * 8;    // A frag K bases {0,8} (+0 / +16 halves)
    const int kbb  = (lane >> 4) * 8;    // B frag K bases {0,8} (+0 / +16 halves)

    auto stage = [&](int buf, int k0) {
#pragma unroll
        for (int i = 0; i < 4; ++i) {
            const int row = srow0 + i * 32;
            async_b128_to_lds(A + (size_t)(blockM + row) * K + k0 + sc4,
                              &Asf[buf][row * LDSF + sc4]);
            async_b128_to_lds(B + (size_t)(blockN + row) * K + k0 + sc4,
                              &Bsf[buf][row * LDSF + sc4]);
        }
    };

    // Convert the landed f32 tile pair to bf16 exactly once: 1 row / thread.
    const int   crow = t & 127;
    auto convert = [&](int buf) {
        const float*  src = (t < 128) ? &Asf[buf][crow * LDSF] : &Bsf[buf][crow * LDSF];
        __bf16*       dst = (t < 128) ? &Asb[crow * LDB]       : &Bsb[crow * LDB];
#pragma unroll
        for (int c = 0; c < BK; c += 8) {
            v4f v0 = *(const v4f*)(src + c);
            v4f v1 = *(const v4f*)(src + c + 4);
            v8bf o;
#pragma unroll
            for (int e = 0; e < 4; ++e) {
                o[e]     = (__bf16)v0[e];
                o[4 + e] = (__bf16)v1[e];
            }
            *(v8bf*)(dst + c) = o;
        }
    };

    // Prologue: stage tile 0.
    stage(0, 0);
    wait_async_all();
    __syncthreads();

    const int nIter = K / BK;
    for (int it = 0; it < nIter; ++it) {
        const int buf = it & 1;

        // Kick off next tile's async copies before any compute on this one.
        if (it + 1 < nIter)
            stage(buf ^ 1, (it + 1) * BK);

        // One-shot f32 -> bf16 conversion of the current tile pair.
        convert(buf);
        __syncthreads();                       // bf16 tile visible to all waves

        // ---- bf16 fragment loads from LDS ----
        // A 16x32 layout: lanes 0-15 hold K {0..7, 16..23}, lanes 16-31 the rest.
        v16bf afrag[2];
#pragma unroll
        for (int i = 0; i < 2; ++i) {
            const __bf16* p = &Asb[(wm * 32 + i * 16 + mrow) * LDB];
            v8bf lo = *(const v8bf*)(p + kha);
            v8bf hi = *(const v8bf*)(p + kha + 16);
            afrag[i] = __builtin_shufflevector(lo, hi,
                0, 1, 2, 3, 4, 5, 6, 7, 8, 9, 10, 11, 12, 13, 14, 15);
        }
        // B 32x16 layout: lanes 0-15 hold K {0..7, 16..23}, lanes 16-31 the rest.
        v16bf bfrag[4];
#pragma unroll
        for (int j = 0; j < 4; ++j) {
            const __bf16* p = &Bsb[(wn * 64 + j * 16 + mrow) * LDB];
            v8bf lo = *(const v8bf*)(p + kbb);
            v8bf hi = *(const v8bf*)(p + kbb + 16);
            bfrag[j] = __builtin_shufflevector(lo, hi,
                0, 1, 2, 3, 4, 5, 6, 7, 8, 9, 10, 11, 12, 13, 14, 15);
        }

        // ---- 8 WMMAs per wave per K step ----
#pragma unroll
        for (int i = 0; i < 2; ++i)
#pragma unroll
            for (int j = 0; j < 4; ++j)
                acc[i][j] = __builtin_amdgcn_wmma_f32_16x16x32_bf16(
                    false, afrag[i], false, bfrag[j],
                    (short)0, acc[i][j], false, false);

        // Next f32 tile must land + all waves done with this bf16 tile.
        wait_async_all();
        __syncthreads();
    }

    // ---- epilogue: bias add + store ----
    // C/D layout: VGPR v -> M = v + 8*(lane>>4), N = lane&15 within 16x16 tile.
#pragma unroll
    for (int j = 0; j < 4; ++j) {
        const int n    = blockN + wn * 64 + j * 16 + (lane & 15);
        const float bn = bias[n];
#pragma unroll
        for (int i = 0; i < 2; ++i) {
            const int mbase = blockM + wm * 32 + i * 16 + 8 * (lane >> 4);
#pragma unroll
            for (int v = 0; v < 8; ++v)
                C[(size_t)(mbase + v) * N + n] = acc[i][j][v] + bn;
        }
    }
}

// ---------------------------------------------------------------------------
// Exact per-column top-k mask over the batch dim, in place.
// 32 adjacent columns per block (lane <-> column => coalesced 128B rows).
// Pass 1: top-byte histogram.  Pass 2: compact threshold-bucket keys to LDS
// (expected ~200/col for Gaussian data; CAP=768).  Refine low 24 bits fully
// in LDS.  Pass 3: mask with exact threshold + tie budget.
// ---------------------------------------------------------------------------
#define TCOLS 32
#define HPAD  257   // 257-word rows: bank = (lane + bin) & 63 -> spread
#define CAP   768

__device__ __forceinline__ unsigned mono_key(float f)
{
    unsigned u = __float_as_uint(f);
    return (u & 0x80000000u) ? ~u : (u | 0x80000000u);
}

__global__ __launch_bounds__(256)
void sae_batch_topk_mask(float* __restrict__ enc /* [BATCH, LATENT] */)
{
    const int col0 = blockIdx.x * TCOLS;
    const int t    = threadIdx.x;
    const int lane = t & 31;
    const int wave = t >> 5;          // 8 waves, each owns rows wave::8

    __shared__ unsigned hist[TCOLS][HPAD];
    __shared__ unsigned cand[TCOLS][CAP];   // threshold-bucket candidate keys
    __shared__ unsigned ccnt[TCOLS];
    __shared__ unsigned pref[TCOLS];
    __shared__ unsigned kk[TCOLS];
    __shared__ int      tie[TCOLS];

    // ---- pass 1: 256-bin histogram of the top key byte -------------------
    for (int i = t; i < TCOLS * 256; i += 256)
        hist[i >> 8][i & 255] = 0;
    if (t < TCOLS) { kk[t] = TOPK; ccnt[t] = 0; }
    __syncthreads();

    for (int r = wave; r < BATCH; r += 8) {
        const unsigned ky = mono_key(enc[(size_t)r * LATENT + col0 + lane]);
        atomicAdd(&hist[lane][ky >> 24], 1u);
    }
    __syncthreads();

    if (t < TCOLS) {
        unsigned k = kk[t];
        int b = 255;
        for (; b > 0; --b) {
            const unsigned c = hist[t][b];
            if (c >= k) break;
            k -= c;
        }
        pref[t] = (unsigned)b << 24;
        kk[t]   = k;                       // rank within threshold bucket
    }
    __syncthreads();

    // ---- pass 2: compact candidate keys (top byte == bucket) --------------
    {
        const unsigned b1 = pref[lane] >> 24;
        for (int r = wave; r < BATCH; r += 8) {
            const unsigned ky = mono_key(enc[(size_t)r * LATENT + col0 + lane]);
            if ((ky >> 24) == b1) {
                const unsigned pos = atomicAdd(&ccnt[lane], 1u);
                if (pos < CAP) cand[lane][pos] = ky;
            }
        }
    }
    __syncthreads();

    // ---- in-LDS refinement of the low 24 bits (8 threads per column) -----
    const int rcol = t & 31;
    const int rsub = t >> 5;
#pragma unroll
    for (int shift = 16; shift >= 0; shift -= 8) {
        for (int i = t; i < TCOLS * 256; i += 256)
            hist[i >> 8][i & 255] = 0;
        __syncthreads();

        const unsigned p = pref[rcol];
        const unsigned nc = ccnt[rcol];
        const unsigned n  = (nc < (unsigned)CAP) ? nc : (unsigned)CAP;
        for (unsigned i = rsub; i < n; i += 8) {
            const unsigned ky = cand[rcol][i];
            if ((ky >> (shift + 8)) == (p >> (shift + 8)))
                atomicAdd(&hist[rcol][(ky >> shift) & 255u], 1u);
        }
        __syncthreads();

        if (t < TCOLS && ccnt[t] <= (unsigned)CAP) {   // exact path
            unsigned k = kk[t];
            int b = 255;
            for (; b > 0; --b) {
                const unsigned c = hist[t][b];
                if (c >= k) break;
                k -= c;
            }
            pref[t] |= (unsigned)b << shift;
            kk[t]    = k;
        }
        __syncthreads();
    }

    // ---- pass 3: apply mask ----------------------------------------------
    if (t < TCOLS) tie[t] = (int)kk[t];   // # elems == threshold to keep
    __syncthreads();

    unsigned th = pref[lane];
    if (ccnt[lane] > (unsigned)CAP)       // degenerate fallback: bucket ceiling
        th |= 0x00FFFFFFu;

    for (int r = wave; r < BATCH; r += 8) {
        const size_t   idx = (size_t)r * LATENT + col0 + lane;
        const unsigned ky  = mono_key(enc[idx]);
        if (ky > th) continue;                       // strictly above: keep
        if (ky == th) {
            const int old = atomicSub(&tie[lane], 1);
            if (old > 0) continue;                   // tie within budget: keep
        }
        enc[idx] = 0.0f;                             // below threshold: zero
    }
}

// ---------------------------------------------------------------------------
extern "C" void kernel_launch(void* const* d_in, const int* in_sizes, int n_in,
                              void* d_out, int out_size, void* d_ws, size_t ws_size,
                              hipStream_t stream)
{
    const float* x     = (const float*)d_in[0];   // [8192, 768]
    const float* enc_w = (const float*)d_in[1];   // [16384, 768]
    const float* enc_b = (const float*)d_in[2];   // [16384]
    const float* dec_w = (const float*)d_in[3];   // [768, 16384]
    const float* dec_b = (const float*)d_in[4];   // [768]

    float* decoded = (float*)d_out;                          // [8192, 768]
    float* sparse  = (float*)d_out + (size_t)BATCH * IN_DIM; // [8192, 16384]

    // GEMM1: encoded = x @ enc_w.T + enc_b  -> written into sparse slot
    sae_gemm_bf16_wmma<<<dim3(LATENT / BN, BATCH / BM), dim3(256), 0, stream>>>(
        x, enc_w, enc_b, sparse, BATCH, LATENT, IN_DIM);

    // Top-k mask over batch dim, in place
    sae_batch_topk_mask<<<dim3(LATENT / TCOLS), dim3(256), 0, stream>>>(sparse);

    // GEMM2: decoded = sparse @ dec_w.T + dec_b
    sae_gemm_bf16_wmma<<<dim3(IN_DIM / BN, BATCH / BM), dim3(256), 0, stream>>>(
        sparse, dec_w, dec_b, decoded, BATCH, IN_DIM, LATENT);
}
